// E53SigmoidGate_45792941310470
// MI455X (gfx1250) — matrix operand order repaired and checked by reading, not compiled, not run on hardware
//
#include <hip/hip_runtime.h>
#include <stdint.h>

#define D 1024
#define TSTEPS 2048
#define BATCH 8
#define MROWS (TSTEPS * BATCH) /* 16384 */
#define LDP (D + 8)            /* padded LDS row: 1032*2=2064 bytes = 129*16, 16B aligned */

typedef __attribute__((ext_vector_type(16))) __bf16 bf16x16;
typedef __attribute__((ext_vector_type(8))) float f32x8;

union FragBF {
    bf16x16 v;
    uint32_t w[8];
    uint4 q[2];
};

__device__ inline uint16_t f2bf(float f) {
    uint32_t u = __float_as_uint(f);
    u += 0x7FFFu + ((u >> 16) & 1u); // round-to-nearest-even
    return (uint16_t)(u >> 16);
}
__device__ inline uint32_t pack2bf(float lo, float hi) {
    return (uint32_t)f2bf(lo) | ((uint32_t)f2bf(hi) << 16);
}
// silu via v_rcp_f32 (avoids full IEEE division sequence)
__device__ inline float silu_f(float x) {
    return x * __builtin_amdgcn_rcpf(1.0f + __expf(-x));
}

__device__ inline f32x8 wmma_bf16(const FragBF& a, const FragBF& b, f32x8 c) {
    // D = A(16x32 bf16) * B(32x16 bf16) + C(16x16 f32)
    return __builtin_amdgcn_wmma_f32_16x16x32_bf16(
        /*neg_a=*/false, a.v, /*neg_b=*/false, b.v,
        /*c_mod=*/(short)0, c, /*reuse_a=*/false, /*reuse_b=*/false);
}

// A fragment from f32 row-major source, converted to bf16 (RNE)
__device__ inline void load_a_f32(FragBF& a, const float* __restrict__ arow, int k0, int hi) {
    const float4* pa0 = (const float4*)(arow + k0 + hi * 8);
    const float4* pa1 = (const float4*)(arow + k0 + 16 + hi * 8);
    float4 x0 = pa0[0], x1 = pa0[1], x2 = pa1[0], x3 = pa1[1];
    a.w[0] = pack2bf(x0.x, x0.y); a.w[1] = pack2bf(x0.z, x0.w);
    a.w[2] = pack2bf(x1.x, x1.y); a.w[3] = pack2bf(x1.z, x1.w);
    a.w[4] = pack2bf(x2.x, x2.y); a.w[5] = pack2bf(x2.z, x2.w);
    a.w[6] = pack2bf(x3.x, x3.y); a.w[7] = pack2bf(x3.z, x3.w);
}
// A fragment from bf16 row-major source
__device__ inline void load_a_bf(FragBF& a, const uint16_t* __restrict__ arow, int k0, int hi) {
    a.q[0] = *(const uint4*)(arow + k0 + hi * 8);
    a.q[1] = *(const uint4*)(arow + k0 + 16 + hi * 8);
}
// B fragment: bf16 k0+kh..+7 and k0+16+kh..+7 of one weight row (column of B); brow includes bn row
__device__ inline void load_b_bf(FragBF& b, const uint16_t* __restrict__ brow, int k0, int kh) {
    b.q[0] = *(const uint4*)(brow + k0 + kh);
    b.q[1] = *(const uint4*)(brow + k0 + 16 + kh);
}

// ---------------------------------------------------------------------------
// 1) spectral power iteration (one 1024-thread workgroup) -> scale scalar
// ---------------------------------------------------------------------------
__device__ inline float block_reduce_sum(float v, float* red) {
    const int d = threadIdx.x;
    red[d] = v;
    __syncthreads();
    for (int s = 512; s > 0; s >>= 1) {
        if (d < s) red[d] += red[d + s];
        __syncthreads();
    }
    float r = red[0];
    __syncthreads();
    return r;
}

__global__ __launch_bounds__(1024) void spectral_kernel(const float* __restrict__ W,
                                                        const float* __restrict__ u_in,
                                                        float* __restrict__ scale_out) {
    __shared__ float su[D], sv[D], red[D];
    const int d = threadIdx.x;
    su[d] = u_in[d];
    __syncthreads();
    float n2 = block_reduce_sum(su[d] * su[d], red);
    su[d] *= rsqrtf(n2);
    __syncthreads();
    for (int it = 0; it < 3; ++it) {
        // v = W^T u   (coalesced: thread d walks column d)
        float acc = 0.f;
        for (int e = 0; e < D; ++e) acc = fmaf(W[(size_t)e * D + d], su[e], acc);
        float nv = sqrtf(block_reduce_sum(acc * acc, red)) + 1e-8f;
        sv[d] = acc / nv;
        __syncthreads();
        // u = W v
        const float* Wr = W + (size_t)d * D;
        float acc2 = 0.f;
        for (int k = 0; k < D; ++k) acc2 = fmaf(Wr[k], sv[k], acc2);
        float nu = sqrtf(block_reduce_sum(acc2 * acc2, red)) + 1e-8f;
        su[d] = acc2 / nu;
        __syncthreads();
    }
    // sigma = | u . (W v) |
    const float* Wr = W + (size_t)d * D;
    float wv = 0.f;
    for (int k = 0; k < D; ++k) wv = fmaf(Wr[k], sv[k], wv);
    float sigma = block_reduce_sum(su[d] * wv, red);
    if (d == 0) scale_out[0] = 0.999f / (fabsf(sigma) + 1e-8f);
}

// ---------------------------------------------------------------------------
// 2) weights -> bf16 (W scaled by spectral scale); row-major = B-fragment layout
// ---------------------------------------------------------------------------
__global__ void cvt_weights_kernel(const float* __restrict__ W, const float* __restrict__ inw,
                                   const float* __restrict__ outw, const float* __restrict__ scale,
                                   uint16_t* __restrict__ Wb, uint16_t* __restrict__ Pin,
                                   uint16_t* __restrict__ Pout) {
    const int i = blockIdx.x * blockDim.x + threadIdx.x;
    if (i >= D * D) return;
    const float s = scale[0];
    Wb[i] = f2bf(W[i] * s);
    Pin[i] = f2bf(inw[i]);
    Pout[i] = f2bf(outw[i]);
}

// ---------------------------------------------------------------------------
// 3) scan init: h buffers (16 padded rows, rows>=8 stay zero), reset barrier
// ---------------------------------------------------------------------------
__global__ void scan_init_kernel(const float* __restrict__ h0, uint16_t* __restrict__ hb0,
                                 uint16_t* __restrict__ hb1, int* __restrict__ bar) {
    const int i = blockIdx.x * blockDim.x + threadIdx.x;
    if (i < 16 * D) {
        const int r = i >> 10;
        hb0[i] = (r < BATCH) ? f2bf(h0[i]) : (uint16_t)0;
        hb1[i] = 0;
    }
    if (i == 0) *bar = 0;
}

// ---------------------------------------------------------------------------
// 4) in_proj GEMM + SiLU:  xproj[t*8+b][e] = silu( sum_d x[b][t][d]*Pin[e][d] )
//    per wave: 32x64 tile (2 A-frags x 4 B-frags = 8 WMMA per K-step),
//    two-stage software pipeline on the K loop.
// ---------------------------------------------------------------------------
__global__ __launch_bounds__(256) void inproj_silu_kernel(const float* __restrict__ x,
                                                          const uint16_t* __restrict__ Pin,
                                                          uint16_t* __restrict__ xproj) {
    const int lane = threadIdx.x & 31;
    const int wid = blockIdx.x * (blockDim.x >> 5) + (threadIdx.x >> 5); // 0..8191
    const int ng = wid & 15;
    const int mt = wid >> 4; // 0..511, tile of 32 rows
    const int n0 = ng * 64;
    const int r = lane & 15;
    const int hi = lane >> 4;
    const int bn = lane & 15;
    const int kh = hi * 8;

    // two A rows (row index == t*8+b)
    const int row0 = mt * 32 + r;
    const int row1 = row0 + 16;
    const float* arow0 = x + ((size_t)(row0 & 7) * TSTEPS + (row0 >> 3)) * D;
    const float* arow1 = x + ((size_t)(row1 & 7) * TSTEPS + (row1 >> 3)) * D;
    const uint16_t* brow[4];
#pragma unroll
    for (int j = 0; j < 4; ++j) brow[j] = Pin + (size_t)(n0 + j * 16 + bn) * D;

    f32x8 acc[8];
#pragma unroll
    for (int i = 0; i < 8; ++i)
#pragma unroll
        for (int v = 0; v < 8; ++v) acc[i][v] = 0.f;

    FragBF aA[2], bA[4], aB[2], bB[4];
    load_a_f32(aA[0], arow0, 0, hi);
    load_a_f32(aA[1], arow1, 0, hi);
#pragma unroll
    for (int j = 0; j < 4; ++j) load_b_bf(bA[j], brow[j], 0, kh);

    for (int k0 = 0; k0 < D; k0 += 64) {
        // stage B loads (k0+32 always < D for D multiple of 64)
        load_a_f32(aB[0], arow0, k0 + 32, hi);
        load_a_f32(aB[1], arow1, k0 + 32, hi);
#pragma unroll
        for (int j = 0; j < 4; ++j) load_b_bf(bB[j], brow[j], k0 + 32, kh);
#pragma unroll
        for (int mi = 0; mi < 2; ++mi)
#pragma unroll
            for (int j = 0; j < 4; ++j) acc[mi * 4 + j] = wmma_bf16(aA[mi], bA[j], acc[mi * 4 + j]);
        if (k0 + 64 < D) {
            load_a_f32(aA[0], arow0, k0 + 64, hi);
            load_a_f32(aA[1], arow1, k0 + 64, hi);
#pragma unroll
            for (int j = 0; j < 4; ++j) load_b_bf(bA[j], brow[j], k0 + 64, kh);
        }
#pragma unroll
        for (int mi = 0; mi < 2; ++mi)
#pragma unroll
            for (int j = 0; j < 4; ++j) acc[mi * 4 + j] = wmma_bf16(aB[mi], bB[j], acc[mi * 4 + j]);
    }

    // store silu(acc) as bf16; C layout: lane holds N=lane&15, VGPR v -> M = v + 8*hi
    const int cn = lane & 15;
#pragma unroll
    for (int mi = 0; mi < 2; ++mi) {
        const int mbase = mt * 32 + mi * 16 + hi * 8;
#pragma unroll
        for (int j = 0; j < 4; ++j)
#pragma unroll
            for (int v = 0; v < 8; ++v)
                xproj[(size_t)(mbase + v) * D + n0 + j * 16 + cn] = f2bf(silu_f(acc[mi * 4 + j][v]));
    }
}

// ---------------------------------------------------------------------------
// 5) persistent scan: h_t = (x'_t + h_{t-1}) @ W_eff^T + b
//    64 blocks x 1 wave; wave owns 16 output columns; W slice lives in LDS;
//    h/x' and LDS B fragment loads double-buffered behind the WMMA chain.
// ---------------------------------------------------------------------------
__global__ __launch_bounds__(32) void scan_kernel(const uint16_t* __restrict__ xproj,
                                                  const uint16_t* __restrict__ Wb,
                                                  const float* __restrict__ bias,
                                                  uint16_t* __restrict__ hb0,
                                                  uint16_t* __restrict__ hb1,
                                                  uint16_t* __restrict__ hs,
                                                  float* __restrict__ hfinal,
                                                  int* __restrict__ bar, int nblocks) {
    __shared__ __align__(16) uint16_t sW[16 * LDP]; // ~33 KB: this wave's 16 W rows
    const int lane = threadIdx.x;
    const int n0 = blockIdx.x * 16;

    for (int idx = lane; idx < 16 * (D / 8); idx += 32) {
        const int rr = idx >> 7;
        const int cc = (idx & 127) * 8;
        *(uint4*)(&sW[rr * LDP + cc]) = *(const uint4*)(Wb + (size_t)(n0 + rr) * D + cc);
    }
    __syncthreads();

    const int r = lane & 15;
    const int hi = lane >> 4;
    const int bn = lane & 15;
    const uint16_t* swrow = &sW[bn * LDP]; // this lane's B column source
    const int kh = hi * 8;
    const float bias_n = bias[n0 + bn];
    f32x8 cinit;
#pragma unroll
    for (int v = 0; v < 8; ++v) cinit[v] = bias_n; // C[m][n] = b[n]

    for (int t = 0; t < TSTEPS; ++t) {
        const uint16_t* hprev = (t & 1) ? hb1 : hb0;
        uint16_t* hnext = (t & 1) ? hb0 : hb1;
        const uint16_t* xp = xproj + (size_t)t * BATCH * D;
        f32x8 c = cinit;

        // A-frag builder: a = bf16(h_prev + x'_t) (rows >= 8 are zero padding)
        auto load_a = [&](FragBF& a, int k0) {
            if (r < BATCH) {
                FragBF h, xv;
                h.q[0] = *(const uint4*)(hprev + (size_t)r * D + k0 + hi * 8);
                h.q[1] = *(const uint4*)(hprev + (size_t)r * D + k0 + 16 + hi * 8);
                xv.q[0] = *(const uint4*)(xp + (size_t)r * D + k0 + hi * 8);
                xv.q[1] = *(const uint4*)(xp + (size_t)r * D + k0 + 16 + hi * 8);
#pragma unroll
                for (int w = 0; w < 8; ++w) {
                    uint32_t o;
                    asm volatile("v_pk_add_bf16 %0, %1, %2" : "=v"(o) : "v"(h.w[w]), "v"(xv.w[w]));
                    a.w[w] = o;
                }
            } else {
#pragma unroll
                for (int w = 0; w < 8; ++w) a.w[w] = 0u;
            }
        };

        FragBF aA, bA, aB, bB;
        load_a(aA, 0);
        load_b_bf(bA, swrow, 0, kh);
        for (int k0 = 0; k0 < D; k0 += 64) {
            load_a(aB, k0 + 32);
            load_b_bf(bB, swrow, k0 + 32, kh);
            c = wmma_bf16(aA, bA, c);
            if (k0 + 64 < D) {
                load_a(aA, k0 + 64);
                load_b_bf(bA, swrow, k0 + 64, kh);
            }
            c = wmma_bf16(aB, bB, c);
        }

        // rows 0..7 (= batch) live in lanes 0..15, VGPRs 0..7
        if (hi == 0) {
#pragma unroll
            for (int v = 0; v < 8; ++v) {
                const float val = c[v];
                hnext[(size_t)v * D + n0 + bn] = f2bf(val);
                hs[((size_t)t * BATCH + v) * D + n0 + bn] = f2bf(silu_f(val));
                if (t == TSTEPS - 1) hfinal[(size_t)v * D + n0 + bn] = val;
            }
        }

        // grid-wide barrier (monotone counter, reset by scan_init each launch)
        __threadfence();
        __syncthreads();
        if (lane == 0) {
            __hip_atomic_fetch_add(bar, 1, __ATOMIC_RELEASE, __HIP_MEMORY_SCOPE_AGENT);
            const int target = nblocks * (t + 1);
            while (__hip_atomic_load(bar, __ATOMIC_ACQUIRE, __HIP_MEMORY_SCOPE_AGENT) < target) {
                __builtin_amdgcn_s_sleep(1);
            }
        }
        __syncthreads();
        __threadfence();
    }
}

// ---------------------------------------------------------------------------
// 6) out_proj GEMM: y[b][t][o] = sum_d hs[t*8+b][d] * Pout[o][d]
//    same 32x64 per-wave tiling + two-stage pipeline, bf16 A.
// ---------------------------------------------------------------------------
__global__ __launch_bounds__(256) void outproj_kernel(const uint16_t* __restrict__ hs,
                                                      const uint16_t* __restrict__ Pout,
                                                      float* __restrict__ y) {
    const int lane = threadIdx.x & 31;
    const int wid = blockIdx.x * (blockDim.x >> 5) + (threadIdx.x >> 5); // 0..8191
    const int ng = wid & 15;
    const int mt = wid >> 4;
    const int n0 = ng * 64;
    const int r = lane & 15;
    const int hi = lane >> 4;
    const int bn = lane & 15;
    const int kh = hi * 8;

    const uint16_t* arow0 = hs + (size_t)(mt * 32 + r) * D;
    const uint16_t* arow1 = hs + (size_t)(mt * 32 + 16 + r) * D;
    const uint16_t* brow[4];
#pragma unroll
    for (int j = 0; j < 4; ++j) brow[j] = Pout + (size_t)(n0 + j * 16 + bn) * D;

    f32x8 acc[8];
#pragma unroll
    for (int i = 0; i < 8; ++i)
#pragma unroll
        for (int v = 0; v < 8; ++v) acc[i][v] = 0.f;

    FragBF aA[2], bA[4], aB[2], bB[4];
    load_a_bf(aA[0], arow0, 0, hi);
    load_a_bf(aA[1], arow1, 0, hi);
#pragma unroll
    for (int j = 0; j < 4; ++j) load_b_bf(bA[j], brow[j], 0, kh);

    for (int k0 = 0; k0 < D; k0 += 64) {
        load_a_bf(aB[0], arow0, k0 + 32, hi);
        load_a_bf(aB[1], arow1, k0 + 32, hi);
#pragma unroll
        for (int j = 0; j < 4; ++j) load_b_bf(bB[j], brow[j], k0 + 32, kh);
#pragma unroll
        for (int mi = 0; mi < 2; ++mi)
#pragma unroll
            for (int j = 0; j < 4; ++j) acc[mi * 4 + j] = wmma_bf16(aA[mi], bA[j], acc[mi * 4 + j]);
        if (k0 + 64 < D) {
            load_a_bf(aA[0], arow0, k0 + 64, hi);
            load_a_bf(aA[1], arow1, k0 + 64, hi);
#pragma unroll
            for (int j = 0; j < 4; ++j) load_b_bf(bA[j], brow[j], k0 + 64, kh);
        }
#pragma unroll
        for (int mi = 0; mi < 2; ++mi)
#pragma unroll
            for (int j = 0; j < 4; ++j) acc[mi * 4 + j] = wmma_bf16(aB[mi], bB[j], acc[mi * 4 + j]);
    }

    const int cn = lane & 15;
#pragma unroll
    for (int mi = 0; mi < 2; ++mi) {
        const int mbase = mt * 32 + mi * 16 + hi * 8;
#pragma unroll
        for (int j = 0; j < 4; ++j)
#pragma unroll
            for (int v = 0; v < 8; ++v) {
                const int row = mbase + v; // == t*8+b
                const int tt = row >> 3;
                const int bb = row & 7;
                y[(size_t)bb * TSTEPS * D + (size_t)tt * D + n0 + j * 16 + cn] = acc[mi * 4 + j][v];
            }
    }
}

// ---------------------------------------------------------------------------
// host: stage the pipeline on `stream` (graph-capture safe, deterministic)
// ---------------------------------------------------------------------------
extern "C" void kernel_launch(void* const* d_in, const int* in_sizes, int n_in,
                              void* d_out, int out_size, void* d_ws, size_t ws_size,
                              hipStream_t stream) {
    (void)in_sizes; (void)n_in; (void)out_size; (void)ws_size;
    const float* x = (const float*)d_in[0];
    const float* h0 = (const float*)d_in[1];
    const float* W = (const float*)d_in[2];
    const float* bias = (const float*)d_in[3];
    const float* u = (const float*)d_in[4];
    const float* inw = (const float*)d_in[5];
    const float* outw = (const float*)d_in[6];
    float* y = (float*)d_out;                       // [B,T,D]
    float* hfinal = y + (size_t)BATCH * TSTEPS * D; // [B,D] appended

    uint8_t* ws = (uint8_t*)d_ws;
    size_t off = 0;
    auto salloc = [&](size_t bytes) -> void* {
        void* p = ws + off;
        off = (off + bytes + 255) & ~(size_t)255;
        return p;
    };
    float* scale = (float*)salloc(sizeof(float));
    int* bar = (int*)salloc(sizeof(int));
    uint16_t* Wb = (uint16_t*)salloc((size_t)D * D * 2);        // 2 MB
    uint16_t* Pin = (uint16_t*)salloc((size_t)D * D * 2);       // 2 MB
    uint16_t* Pout = (uint16_t*)salloc((size_t)D * D * 2);      // 2 MB
    uint16_t* xproj = (uint16_t*)salloc((size_t)MROWS * D * 2); // 32 MB, [t*8+b][D]
    uint16_t* hs = (uint16_t*)salloc((size_t)MROWS * D * 2);    // 32 MB, silu(h)
    uint16_t* hb0 = (uint16_t*)salloc((size_t)16 * D * 2);
    uint16_t* hb1 = (uint16_t*)salloc((size_t)16 * D * 2);

    hipLaunchKernelGGL(spectral_kernel, dim3(1), dim3(D), 0, stream, W, u, scale);
    hipLaunchKernelGGL(cvt_weights_kernel, dim3((D * D + 255) / 256), dim3(256), 0, stream,
                       W, inw, outw, scale, Wb, Pin, Pout);
    hipLaunchKernelGGL(scan_init_kernel, dim3((16 * D + 255) / 256), dim3(256), 0, stream,
                       h0, hb0, hb1, bar);
    hipLaunchKernelGGL(inproj_silu_kernel, dim3(1024), dim3(256), 0, stream, x, Pin, xproj);
    hipLaunchKernelGGL(scan_kernel, dim3(64), dim3(32), 0, stream,
                       xproj, Wb, bias, hb0, hb1, hs, hfinal, bar, 64);
    hipLaunchKernelGGL(outproj_kernel, dim3(1024), dim3(256), 0, stream, hs, Pout, y);
}
